// SqrllResid_52621939311019
// MI455X (gfx1250) — matrix operand order repaired and checked by reading, not compiled, not run on hardware
//
#include <hip/hip_runtime.h>
#include <hip/hip_bf16.h>
#include <math.h>

// ---------------------------------------------------------------------------
// Problem constants (match reference): B=4, T=4096, D=1024, M=1024
// ---------------------------------------------------------------------------
#define CB 4
#define CT 4096
#define CD 1024
#define CM 1024
#define CN (CB * CT)          // 16384 rows
#define NSEG 32               // recurrence segments
#define SEGLEN (CT / NSEG)    // 128 steps per segment

// All GEMMs in this model are [CN x 1024] x [1024 x 1024]^T
#define GK 1024               // reduction dim (compile-time)
#define GMO 1024              // output width  (compile-time)
#define TK 32                 // K-step per LDS tile
#define NK (GK / TK)          // 32 K-iterations
#define SSTR 40               // padded LDS row stride (halves) -> no bank conflicts

typedef __attribute__((ext_vector_type(16))) __bf16 v16bf;
typedef __attribute__((ext_vector_type(8)))  __bf16 v8bf;
typedef __attribute__((ext_vector_type(8)))  float  v8f;

// CDNA5 async global->LDS path (ASYNCcnt), guarded so compile never breaks.
// Probe result: param0 = AS1 int4* (non-const), param1 = AS3 int4*, then
// (imm offset, imm cpol).
#if defined(__AMDGCN__) && __has_builtin(__builtin_amdgcn_global_load_async_to_lds_b128) && __has_builtin(__builtin_amdgcn_s_wait_asynccnt)
#define HAVE_ASYNC_LDS 1
typedef int async_v4i __attribute__((vector_size(16)));
#define ASYNC_B128(gp, lp)                                                    \
    __builtin_amdgcn_global_load_async_to_lds_b128(                           \
        (__attribute__((address_space(1))) async_v4i*)(gp),                   \
        (__attribute__((address_space(3))) async_v4i*)(lp), 0, 0)
#else
#define HAVE_ASYNC_LDS 0
#endif

static __device__ __forceinline__ float sigm(float v) {
    return 1.0f / (1.0f + __expf(-v));
}

// ---------------------------------------------------------------------------
// f32 -> bf16 conversion (weights)
// ---------------------------------------------------------------------------
__global__ __launch_bounds__(256) void f32_to_bf16_k(const float* __restrict__ s,
                                                     __bf16* __restrict__ d, int n) {
    int i = (blockIdx.x * 256 + threadIdx.x) * 4;
    if (i + 3 < n) {
        d[i + 0] = (__bf16)s[i + 0];
        d[i + 1] = (__bf16)s[i + 1];
        d[i + 2] = (__bf16)s[i + 2];
        d[i + 3] = (__bf16)s[i + 3];
    }
}

// ---------------------------------------------------------------------------
// RMSNorm: one 256-thread block per row of D=1024, emits bf16
// ---------------------------------------------------------------------------
__global__ __launch_bounds__(256) void rmsnorm_bf16_k(const float* __restrict__ x,
                                                      const float* __restrict__ w,
                                                      __bf16* __restrict__ y) {
    const int row = blockIdx.x;
    const int tid = threadIdx.x;
    const float* xr = x + (size_t)row * CD;
    float v[4];
    float ss = 0.0f;
#pragma unroll
    for (int i = 0; i < 4; ++i) {
        v[i] = xr[tid + i * 256];
        ss += v[i] * v[i];
    }
    __shared__ float red[256];
    red[tid] = ss;
    for (int s = 128; s > 0; s >>= 1) {
        __syncthreads();
        if (tid < s) red[tid] += red[tid + s];
    }
    __syncthreads();
    const float scale = rsqrtf(red[0] * (1.0f / (float)CD) + 1e-6f);
#pragma unroll
    for (int i = 0; i < 4; ++i) {
        const int c = tid + i * 256;
        y[(size_t)row * CD + c] = (__bf16)(v[i] * scale * w[c]);
    }
}

// ---------------------------------------------------------------------------
// WMMA GEMM: C[N,1024](f32) = epi( A[N,1024](bf16) x W[1024,1024](bf16)^T )
// Block tile 128x128, K-step 32, 256 threads = 8 waves (2x4); each wave does
// 64x32 = 4x2 tiles of v_wmma_f32_16x16x32_bf16.  Double-buffered LDS with a
// software-pipelined K loop; tiles staged either by GLOBAL_LOAD_ASYNC_TO_LDS
// (ASYNCcnt path) or by a register double-buffer fallback.
// EPI: 0 = raw, 1 = sigmoid(acc + bias[n]), 2 = acc + resid[row,n]
// ---------------------------------------------------------------------------
template <int EPI>
__global__ __launch_bounds__(256) void gemm_bf16_k(const __bf16* __restrict__ A,
                                                   const __bf16* __restrict__ W,
                                                   const float* __restrict__ bias,
                                                   const float* __restrict__ resid,
                                                   float* __restrict__ C) {
    __shared__ __bf16 sA[2][128 * SSTR];
    __shared__ __bf16 sB[2][128 * SSTR];

    const int tid  = threadIdx.x;
    const int lane = tid & 31;
    const int wave = tid >> 5;
    const int wr   = wave >> 2;   // 0..1 : wave row-group
    const int wc   = wave & 3;    // 0..3 : wave col-group
    const int rowBase = blockIdx.y * 128;
    const int colBase = blockIdx.x * 128;

    const int ldRow = tid >> 1;          // 0..127
    const int ldCol = (tid & 1) * 16;    // 0 / 16
    const int m16 = lane & 15;
    const int hi  = lane >> 4;           // lane half
    const int kbA = hi * 8;              // A-frag K base (ISA 16-bit A layout)
    const int kbB = hi * 8;              // B-frag K base (ISA 16-bit B layout)

    const __bf16* aRow = A + (size_t)(rowBase + ldRow) * GK + ldCol;
    const __bf16* wRow = W + (size_t)(colBase + ldRow) * GK + ldCol;
    const int ldsOff = ldRow * SSTR + ldCol;

    const v8f vzero = {0.f, 0.f, 0.f, 0.f, 0.f, 0.f, 0.f, 0.f};
    v8f acc[4][2];
#pragma unroll
    for (int i = 0; i < 4; ++i)
#pragma unroll
        for (int j = 0; j < 2; ++j) acc[i][j] = vzero;

    // fragment load + 8x WMMA from LDS buffer b
    auto computeTile = [&](int b) {
        v16bf af[4];
#pragma unroll
        for (int rt = 0; rt < 4; ++rt) {
            const __bf16* p = &sA[b][(wr * 64 + rt * 16 + m16) * SSTR];
            v8bf c0 = *(const v8bf*)(p + kbA);
            v8bf c1 = *(const v8bf*)(p + kbA + 16);
#pragma unroll
            for (int e = 0; e < 8; ++e) { af[rt][e] = c0[e]; af[rt][e + 8] = c1[e]; }
        }
        v16bf bfr[2];
#pragma unroll
        for (int ct = 0; ct < 2; ++ct) {
            const __bf16* p = &sB[b][(wc * 32 + ct * 16 + m16) * SSTR + kbB];
            v8bf d0 = *(const v8bf*)p;
            v8bf d1 = *(const v8bf*)(p + 16);
#pragma unroll
            for (int e = 0; e < 8; ++e) { bfr[ct][e] = d0[e]; bfr[ct][e + 8] = d1[e]; }
        }
#pragma unroll
        for (int rt = 0; rt < 4; ++rt)
#pragma unroll
            for (int ct = 0; ct < 2; ++ct)
                acc[rt][ct] = __builtin_amdgcn_wmma_f32_16x16x32_bf16(
                    false, af[rt], false, bfr[ct], (short)0, acc[rt][ct],
                    false, false);
    };

#if HAVE_ASYNC_LDS
    // ---- ASYNCcnt-pipelined path: tiles go straight from L2/HBM to LDS ----
    auto issueTile = [&](int kk, int b) {
        const __bf16* aSrc = aRow + kk * TK;
        const __bf16* bSrc = wRow + kk * TK;
        ASYNC_B128(aSrc,     &sA[b][ldsOff]);
        ASYNC_B128(aSrc + 8, &sA[b][ldsOff + 8]);
        ASYNC_B128(bSrc,     &sB[b][ldsOff]);
        ASYNC_B128(bSrc + 8, &sB[b][ldsOff + 8]);
    };
    issueTile(0, 0);
#pragma unroll 1
    for (int kk = 0; kk < NK; ++kk) {
        __builtin_amdgcn_s_wait_asynccnt(0);   // own loads for tile kk landed
        __syncthreads();                       // everyone's landed; prev compute done
        if (kk + 1 < NK) issueTile(kk + 1, (kk + 1) & 1);  // overlaps compute below
        computeTile(kk & 1);
    }
#else
    // ---- fallback: register double-buffer (global->VGPR->LDS), pipelined ----
    v8bf rA0, rA1, rB0, rB1;
    auto loadRegs = [&](int kk) {
        const __bf16* aSrc = aRow + kk * TK;
        const __bf16* bSrc = wRow + kk * TK;
        rA0 = *(const v8bf*)aSrc; rA1 = *(const v8bf*)(aSrc + 8);
        rB0 = *(const v8bf*)bSrc; rB1 = *(const v8bf*)(bSrc + 8);
    };
    auto storeRegs = [&](int b) {
        *(v8bf*)&sA[b][ldsOff]     = rA0;
        *(v8bf*)&sA[b][ldsOff + 8] = rA1;
        *(v8bf*)&sB[b][ldsOff]     = rB0;
        *(v8bf*)&sB[b][ldsOff + 8] = rB1;
    };
    loadRegs(0);
    storeRegs(0);
#pragma unroll 1
    for (int kk = 0; kk < NK; ++kk) {
        __syncthreads();                       // tile kk stores visible everywhere
        if (kk + 1 < NK) loadRegs(kk + 1);     // overlaps compute below
        computeTile(kk & 1);
        if (kk + 1 < NK) storeRegs((kk + 1) & 1);
    }
#endif

    // --- epilogue: C/D layout -> VGPR j holds row (j + 8*hi), col = lane%16 ---
#pragma unroll
    for (int rt = 0; rt < 4; ++rt) {
#pragma unroll
        for (int ct = 0; ct < 2; ++ct) {
            const int n = colBase + wc * 32 + ct * 16 + m16;
            const int rbase = rowBase + wr * 64 + rt * 16 + hi * 8;
            float bv = 0.0f;
            if (EPI == 1) bv = bias[n];
#pragma unroll
            for (int j = 0; j < 8; ++j) {
                float v = acc[rt][ct][j];
                const size_t off = (size_t)(rbase + j) * GMO + n;
                if (EPI == 1) v = sigm(v + bv);
                else if (EPI == 2) v += resid[off];
                C[off] = v;
            }
        }
    }
}

// ---------------------------------------------------------------------------
// Recurrence h_t = gi*ig + r*h_{t-1}: 3-pass chunked scan over T
// ---------------------------------------------------------------------------
__global__ __launch_bounds__(256) void rec_partial_k(const float* __restrict__ gi,
                                                     const float* __restrict__ ig,
                                                     const float* __restrict__ r,
                                                     float* __restrict__ Rp,
                                                     float* __restrict__ Hp) {
    const int idx = blockIdx.x * 256 + threadIdx.x;   // b*NSEG*M + seg*M + m
    const int m = idx % CM;
    const size_t base = (size_t)(idx / CM) * SEGLEN * CM + m;
    float R = 1.0f, H = 0.0f;
#pragma unroll 4
    for (int t = 0; t < SEGLEN; ++t) {
        const size_t o = base + (size_t)t * CM;
        const float rv = r[o];
        H = gi[o] * ig[o] + rv * H;
        R *= rv;
    }
    Rp[idx] = R;
    Hp[idx] = H;
}

__global__ __launch_bounds__(256) void rec_scan_k(const float* __restrict__ Rp,
                                                  const float* __restrict__ Hp,
                                                  const float* __restrict__ mem,
                                                  float* __restrict__ h0) {
    const int ch = blockIdx.x * 256 + threadIdx.x;    // b*M + m
    const int b = ch / CM, m = ch % CM;
    float h = mem[ch];
#pragma unroll
    for (int seg = 0; seg < NSEG; ++seg) {
        const int i = (b * NSEG + seg) * CM + m;
        h0[i] = h;
        h = Hp[i] + Rp[i] * h;
    }
}

__global__ __launch_bounds__(256) void rec_emit_k(const float* __restrict__ gi,
                                                  const float* __restrict__ ig,
                                                  const float* __restrict__ r,
                                                  const float* __restrict__ og,
                                                  const float* __restrict__ h0,
                                                  __bf16* __restrict__ s,
                                                  float* __restrict__ memout) {
    const int idx = blockIdx.x * 256 + threadIdx.x;
    const int m = idx % CM;
    const size_t base = (size_t)(idx / CM) * SEGLEN * CM + m;
    float h = h0[idx];
#pragma unroll 4
    for (int t = 0; t < SEGLEN; ++t) {
        const size_t o = base + (size_t)t * CM;
        h = gi[o] * ig[o] + r[o] * h;
        s[o] = (__bf16)((h / (1.0f + fabsf(h))) * og[o]);
    }
    if (((idx / CM) % NSEG) == NSEG - 1)
        memout[(idx / (NSEG * CM)) * CM + m] = h;
}

// ff = gi * sig (sig already sigmoided by GEMM epilogue) -> bf16
__global__ __launch_bounds__(256) void gate_mul_bf16_k(const float* __restrict__ a,
                                                       const float* __restrict__ g,
                                                       __bf16* __restrict__ d) {
    const size_t i = ((size_t)blockIdx.x * 256 + threadIdx.x) * 4;
#pragma unroll
    for (int e = 0; e < 4; ++e) d[i + e] = (__bf16)(a[i + e] * g[i + e]);
}

// ---------------------------------------------------------------------------
// Host-side orchestration
// ---------------------------------------------------------------------------
extern "C" void kernel_launch(void* const* d_in, const int* in_sizes, int n_in,
                              void* d_out, int out_size, void* d_ws, size_t ws_size,
                              hipStream_t stream) {
    (void)in_sizes; (void)n_in; (void)out_size; (void)ws_size;
    const float* x       = (const float*)d_in[0];
    const float* mem     = (const float*)d_in[1];
    const float* norm_w  = (const float*)d_in[2];
    const float* wr_w    = (const float*)d_in[3];
    const float* wr_b    = (const float*)d_in[4];
    const float* wi_w    = (const float*)d_in[5];
    const float* wig_w   = (const float*)d_in[6];
    const float* wig_b   = (const float*)d_in[7];
    const float* wog_w   = (const float*)d_in[8];
    const float* wog_b   = (const float*)d_in[9];
    const float* wo_w    = (const float*)d_in[10];
    const float* ffnorm_w= (const float*)d_in[11];
    const float* ffn_wi_w= (const float*)d_in[12];
    const float* ffn_wg_w= (const float*)d_in[13];
    const float* ffn_wg_b= (const float*)d_in[14];
    const float* ffn_wo_w= (const float*)d_in[15];
    float* out = (float*)d_out;

    char* ws = (char*)d_ws;
    const size_t MB = (size_t)1 << 20;
    __bf16* wbf   = (__bf16*)ws;                       // 8 x 2MB bf16 weights
    __bf16* ybuf  = (__bf16*)(ws + 16 * MB);           // 32MB: y / s / z / ff
    float*  g_r   = (float*)(ws + 48 * MB);            // 64MB (reused as x1)
    float*  g_i   = (float*)(ws + 112 * MB);           // 64MB
    float*  g_ig  = (float*)(ws + 176 * MB);           // 64MB
    float*  g_og  = (float*)(ws + 240 * MB);           // 64MB
    float*  Rp    = (float*)(ws + 304 * MB);           // 512KB
    float*  Hp    = Rp + CB * NSEG * CM;               // 512KB
    float*  h0    = Hp + CB * NSEG * CM;               // 512KB
    float*  x1    = g_r;                               // reuse after recurrence

    const int WN = CM * CD;   // 1M elements per weight
    auto wslot = [&](int i) { return wbf + (size_t)i * WN; };

    // weights -> bf16
    f32_to_bf16_k<<<WN / 1024, 256, 0, stream>>>(wr_w,     wslot(0), WN);
    f32_to_bf16_k<<<WN / 1024, 256, 0, stream>>>(wi_w,     wslot(1), WN);
    f32_to_bf16_k<<<WN / 1024, 256, 0, stream>>>(wig_w,    wslot(2), WN);
    f32_to_bf16_k<<<WN / 1024, 256, 0, stream>>>(wog_w,    wslot(3), WN);
    f32_to_bf16_k<<<WN / 1024, 256, 0, stream>>>(wo_w,     wslot(4), WN);
    f32_to_bf16_k<<<WN / 1024, 256, 0, stream>>>(ffn_wi_w, wslot(5), WN);
    f32_to_bf16_k<<<WN / 1024, 256, 0, stream>>>(ffn_wg_w, wslot(6), WN);
    f32_to_bf16_k<<<WN / 1024, 256, 0, stream>>>(ffn_wo_w, wslot(7), WN);

    // y = rmsnorm(x) (bf16)
    rmsnorm_bf16_k<<<CN, 256, 0, stream>>>(x, norm_w, ybuf);

    // 4 gate GEMMs
    dim3 gg(GMO / 128, CN / 128, 1);
    gemm_bf16_k<1><<<gg, 256, 0, stream>>>(ybuf, wslot(0), wr_b,    nullptr, g_r);
    gemm_bf16_k<0><<<gg, 256, 0, stream>>>(ybuf, wslot(1), nullptr, nullptr, g_i);
    gemm_bf16_k<1><<<gg, 256, 0, stream>>>(ybuf, wslot(2), wig_b,   nullptr, g_ig);
    gemm_bf16_k<1><<<gg, 256, 0, stream>>>(ybuf, wslot(3), wog_b,   nullptr, g_og);

    // chunked linear recurrence; s (bf16) overwrites ybuf; mem_out -> d_out tail
    rec_partial_k<<<(CB * NSEG * CM) / 256, 256, 0, stream>>>(g_i, g_ig, g_r, Rp, Hp);
    rec_scan_k<<<(CB * CM) / 256, 256, 0, stream>>>(Rp, Hp, mem, h0);
    rec_emit_k<<<(CB * NSEG * CM) / 256, 256, 0, stream>>>(g_i, g_ig, g_r, g_og, h0,
                                                           ybuf, out + (size_t)CN * CD);

    // x1 = x + s @ wo^T
    gemm_bf16_k<2><<<gg, 256, 0, stream>>>(ybuf, wslot(4), nullptr, x, x1);

    // FFN branch: z = rmsnorm(x1); ff = (z@wi^T) * sigmoid(z@wg^T + b)
    rmsnorm_bf16_k<<<CN, 256, 0, stream>>>(x1, ffnorm_w, ybuf);
    gemm_bf16_k<0><<<gg, 256, 0, stream>>>(ybuf, wslot(5), nullptr,  nullptr, g_i);
    gemm_bf16_k<1><<<gg, 256, 0, stream>>>(ybuf, wslot(6), ffn_wg_b, nullptr, g_ig);
    gate_mul_bf16_k<<<((size_t)CN * CM) / 1024, 256, 0, stream>>>(g_i, g_ig, ybuf);

    // out = x1 + ff @ ffn_wo^T
    gemm_bf16_k<2><<<gg, 256, 0, stream>>>(ybuf, wslot(7), nullptr, x1, out);
}
